// TransformerEncoderLayer_28879360098514
// MI455X (gfx1250) — compile-verified, matrix operand order checked by reading, not run on hardware
//
#include <hip/hip_runtime.h>
#include <hip/hip_bf16.h>
#include <stdint.h>

// ---------------------------------------------------------------------------
// Types for CDNA5 WMMA (wave32): bf16 A/B fragments (16 x bf16 = 8 VGPRs),
// f32 C/D accumulator (8 VGPRs).
// ---------------------------------------------------------------------------
typedef __attribute__((ext_vector_type(16))) __bf16        bf16x16;
typedef __attribute__((ext_vector_type(8)))  float         f32x8;
typedef __attribute__((ext_vector_type(4)))  unsigned int  u32x4;
typedef __attribute__((ext_vector_type(8)))  int           i32x8;
typedef __attribute__((ext_vector_type(4)))  int           i32x4;

// ---------------------------------------------------------------------------
// Tensor Data Mover: 2D tile (tile_w x tile_h elements of f32, row stride
// `row_stride` elements) from global memory -> LDS at byte offset lds_addr.
// D# packing per CDNA5 ISA ch.8:
//   group0: [1:0]=count=1, [63:32]=lds_addr, [120:64]=global byte addr,
//           [127:126]=type=2
//   group1: [17:16]=data_size (2 -> 4B), [79:48]=tensor_dim0, [111:80]=
//           tensor_dim1, [127:112]=tile_dim0, [143:128]=tile_dim1,
//           [207:160]=tensor_dim0_stride
// Tensor dims are set equal to the tile (tiles are always fully in-bounds).
// ---------------------------------------------------------------------------
__device__ __forceinline__ void tdm_load_2d_f32(unsigned lds_addr,
                                                const float* gptr,
                                                unsigned tile_w, unsigned tile_h,
                                                unsigned long long row_stride) {
    const unsigned long long ga = (unsigned long long)(uintptr_t)gptr;
    u32x4 g0;
    g0[0] = 1u;                                        // count=1 (valid user D#)
    g0[1] = lds_addr;                                  // LDS byte address
    g0[2] = (unsigned)(ga & 0xFFFFFFFFu);              // global_addr[31:0]
    g0[3] = (unsigned)((ga >> 32) & 0x1FFFFFFu)        // global_addr[56:32]
          | (2u << 30);                                // type = 2 ("image")

    i32x8 g1;
    g1[0] = (int)(2u << 16);                                           // data_size=4B
    g1[1] = (int)((tile_w & 0xFFFFu) << 16);                           // tensor_dim0 lo
    g1[2] = (int)(((tile_w >> 16) & 0xFFFFu) |
                  ((tile_h & 0xFFFFu) << 16));                         // dim0 hi | dim1 lo
    g1[3] = (int)(((tile_h >> 16) & 0xFFFFu) |
                  ((tile_w & 0xFFFFu) << 16));                         // dim1 hi | tile_dim0
    g1[4] = (int)(tile_h & 0xFFFFu);                                   // tile_dim1 (tile_dim2=0)
    g1[5] = (int)(row_stride & 0xFFFFFFFFu);                           // dim0_stride lo
    g1[6] = (int)((row_stride >> 32) & 0xFFFFu);                       // dim0_stride hi
    g1[7] = 0;

    const i32x4 z4 = {0, 0, 0, 0};
#if __has_include(<hip/amd_detail/amd_gfx1250_TDM.h>)
    const i32x8 z8 = {0, 0, 0, 0, 0, 0, 0, 0};
    __builtin_amdgcn_tensor_load_to_lds(g0, g1, z4, z4, z8, 0);   // 6-arg toolchain
#else
    __builtin_amdgcn_tensor_load_to_lds(g0, g1, z4, z4, 0);       // ROCm 7.2 (5-arg)
#endif
}

__device__ __forceinline__ unsigned lds_byte_addr(const void* p) {
    // Generic pointers to LDS carry the byte offset in their low 32 bits.
    return (unsigned)(uintptr_t)p;
}

// ---------------------------------------------------------------------------
// Fragment builders from f32 tiles staged in LDS (ISA 7.12.2 layouts).
//   lane&15 -> M (or N); lane>>4 -> K group (0: K 0-7/16-23, 1: K 8-15/24-31)
// Row variant: tile[r][k] row-major with row length ld (A, weight-style B).
// Col variant: tile[k][n] row-major with row length ld (k-major B, e.g. V).
// Plain (__bf16) casts let clang emit native cvt ops.
// ---------------------------------------------------------------------------
__device__ __forceinline__ bf16x16 frag_lds_row(const float* t, int ld, int r0, int lane) {
    const int m = r0 + (lane & 15);
    const int g = (lane >> 4) & 1;
    const float* p = t + m * ld + g * 8;
    bf16x16 f;
#pragma unroll
    for (int i = 0; i < 8; ++i) f[i]     = (__bf16)p[i];
#pragma unroll
    for (int i = 0; i < 8; ++i) f[8 + i] = (__bf16)p[16 + i];
    return f;
}
__device__ __forceinline__ bf16x16 frag_lds_col(const float* t, int ld, int n0, int lane) {
    const int n = n0 + (lane & 15);
    const int g = (lane >> 4) & 1;
    const float* p = t + (g * 8) * ld + n;
    bf16x16 f;
#pragma unroll
    for (int i = 0; i < 8; ++i) f[i]     = (__bf16)p[i * ld];
#pragma unroll
    for (int i = 0; i < 8; ++i) f[8 + i] = (__bf16)p[(16 + i) * ld];
    return f;
}

// ---------------------------------------------------------------------------
// Generic batched GEMM:  C = alpha * A@B + bias + residual [,ReLU]
//   A[m][k] at Abase + m*lda + k     (Abase = A + bz*sAb + hz*sAh)
//   bLayout==0: B[n][k] at Bbase + n*ldb + k   (weight-style W[N,K])
//   bLayout==1: B[k][n] at Bbase + k*ldb + n   (k-major, e.g. V matrix)
//   C[m][n] at Cbase + m*ldc + n ; residual (if non-null) shares C indexing.
// 256 threads = 8 waves; block tile 128(M) x 64(N); each wave owns a 32x32
// sub-tile (2x2 v_wmma_f32_16x16x32_bf16).  K-step 32, TDM double-buffered
// through LDS, synchronized with s_wait_tensorcnt + barrier.
// Requires: M%128==0, N%64==0, K%32==0.  EXEC all-ones at every WMMA.
// ---------------------------------------------------------------------------
__global__ __launch_bounds__(256)
void gemm_bf16_wmma(const float* __restrict__ A, long long sAb, long long sAh, int lda,
                    const float* __restrict__ B, long long sBb, long long sBh, int ldb,
                    float* __restrict__ C, long long sCb, long long sCh, int ldc,
                    const float* __restrict__ bias, const float* __restrict__ residual,
                    int K, float alpha, int batchH, int bLayout, int doRelu) {
    __shared__ float Al[2][128 * 32];   // A tile, [128 rows][32 k] f32
    __shared__ float Bl[2][64 * 32];    // B tile: [64][32] (layout0) or [32][64] (layout1)

    const int tid  = threadIdx.x;
    const int lane = tid & 31;
    const int wave = tid >> 5;
    const int wm   = wave & 3;          // M sub-tile 0..3
    const int wn   = wave >> 2;         // N sub-tile 0..1

    const int z  = blockIdx.z;
    const int bz = z / batchH;
    const int hz = z % batchH;

    const float* Ab = A + (size_t)(bz * sAb + hz * sAh);
    const float* Bb = B + (size_t)(bz * sBb + hz * sBh);
    float*       Cb = C + (size_t)(bz * sCb + hz * sCh);
    const float* Rb = residual ? residual + (size_t)(bz * sCb + hz * sCh) : nullptr;

    const int m0 = blockIdx.x * 128;
    const int n0 = blockIdx.y * 64;

    // ---- prologue: DMA first K-slice into buffer 0 --------------------------
    if (wave == 0) {
        tdm_load_2d_f32(lds_byte_addr(&Al[0][0]), Ab + (size_t)m0 * lda, 32, 128, (unsigned long long)lda);
        if (bLayout == 0)
            tdm_load_2d_f32(lds_byte_addr(&Bl[0][0]), Bb + (size_t)n0 * ldb, 32, 64, (unsigned long long)ldb);
        else
            tdm_load_2d_f32(lds_byte_addr(&Bl[0][0]), Bb + n0, 64, 32, (unsigned long long)ldb);
        __builtin_amdgcn_s_wait_tensorcnt(0);
    }
    __syncthreads();

    f32x8 acc00 = {}, acc01 = {}, acc10 = {}, acc11 = {};

    int cur = 0;
    for (int k0 = 0; k0 < K; k0 += 32) {
        const int nxt = cur ^ 1;
        const bool more = (k0 + 32) < K;

        // ---- async prefetch next K-slice into the other buffer -------------
        if (wave == 0 && more) {
            const int kn = k0 + 32;
            tdm_load_2d_f32(lds_byte_addr(&Al[nxt][0]),
                            Ab + (size_t)m0 * lda + kn, 32, 128, (unsigned long long)lda);
            if (bLayout == 0)
                tdm_load_2d_f32(lds_byte_addr(&Bl[nxt][0]),
                                Bb + (size_t)n0 * ldb + kn, 32, 64, (unsigned long long)ldb);
            else
                tdm_load_2d_f32(lds_byte_addr(&Bl[nxt][0]),
                                Bb + (size_t)kn * ldb + n0, 64, 32, (unsigned long long)ldb);
        }

        // ---- compute on current buffer --------------------------------------
        const float* At = &Al[cur][0];
        const float* Bt = &Bl[cur][0];

        bf16x16 a0 = frag_lds_row(At, 32, wm * 32,      lane);
        bf16x16 a1 = frag_lds_row(At, 32, wm * 32 + 16, lane);
        bf16x16 b0, b1;
        if (bLayout == 0) {
            b0 = frag_lds_row(Bt, 32, wn * 32,      lane);
            b1 = frag_lds_row(Bt, 32, wn * 32 + 16, lane);
        } else {
            b0 = frag_lds_col(Bt, 64, wn * 32,      lane);
            b1 = frag_lds_col(Bt, 64, wn * 32 + 16, lane);
        }

        acc00 = __builtin_amdgcn_wmma_f32_16x16x32_bf16(false, a0, false, b0, (short)0, acc00, false, false);
        acc01 = __builtin_amdgcn_wmma_f32_16x16x32_bf16(false, a0, false, b1, (short)0, acc01, false, false);
        acc10 = __builtin_amdgcn_wmma_f32_16x16x32_bf16(false, a1, false, b0, (short)0, acc10, false, false);
        acc11 = __builtin_amdgcn_wmma_f32_16x16x32_bf16(false, a1, false, b1, (short)0, acc11, false, false);

        if (wave == 0 && more) __builtin_amdgcn_s_wait_tensorcnt(0);
        __syncthreads();
        cur = nxt;
    }

    // ---- epilogue.  D layout: lane<16 -> N=lane, M = vgpr r (+8 lane>=16) ---
    const int nc = lane & 15;
    const int g  = (lane >> 4) & 1;
    const int mbase = m0 + wm * 32;
    const int nbase = n0 + wn * 32;
    f32x8 accs[2][2] = {{acc00, acc01}, {acc10, acc11}};

#pragma unroll
    for (int ti = 0; ti < 2; ++ti) {
#pragma unroll
        for (int tj = 0; tj < 2; ++tj) {
            const int ncol = nbase + tj * 16 + nc;
            const float bv = bias ? bias[ncol] : 0.0f;
#pragma unroll
            for (int r = 0; r < 8; ++r) {
                const int mrow = mbase + ti * 16 + g * 8 + r;
                const size_t idx = (size_t)mrow * ldc + ncol;
                float v = alpha * accs[ti][tj][r] + bv;
                if (Rb)     v += Rb[idx];
                if (doRelu) v = fmaxf(v, 0.0f);
                Cb[idx] = v;
            }
        }
    }
}

// ---------------------------------------------------------------------------
// Block reductions (wave32 shuffles + LDS), blockDim.x == 256.
// ---------------------------------------------------------------------------
__device__ __forceinline__ float warp_sum(float v) {
#pragma unroll
    for (int o = 16; o > 0; o >>= 1) v += __shfl_xor(v, o, 32);
    return v;
}
__device__ __forceinline__ float warp_max(float v) {
#pragma unroll
    for (int o = 16; o > 0; o >>= 1) v = fmaxf(v, __shfl_xor(v, o, 32));
    return v;
}
__device__ __forceinline__ float block_sum(float v, float* sm) {
    const int lane = threadIdx.x & 31, w = threadIdx.x >> 5;
    v = warp_sum(v);
    if (lane == 0) sm[w] = v;
    __syncthreads();
    if (w == 0) {
        float t = (lane < (int)(blockDim.x >> 5)) ? sm[lane] : 0.0f;
        t = warp_sum(t);
        if (lane == 0) sm[0] = t;
    }
    __syncthreads();
    float r = sm[0];
    __syncthreads();
    return r;
}
__device__ __forceinline__ float block_max(float v, float* sm) {
    const int lane = threadIdx.x & 31, w = threadIdx.x >> 5;
    v = warp_max(v);
    if (lane == 0) sm[w] = v;
    __syncthreads();
    if (w == 0) {
        float t = (lane < (int)(blockDim.x >> 5)) ? sm[lane] : -3.4e38f;
        t = warp_max(t);
        if (lane == 0) sm[0] = t;
    }
    __syncthreads();
    float r = sm[0];
    __syncthreads();
    return r;
}

// ---------------------------------------------------------------------------
// Sparsemax over rows of length 1024, in place.  Row-max + 32-step bisection
// on sum(max(z-tau,0))==1 narrows the support set (uniform branches), then an
// exact refine tau=(sum_support-1)/k gives the closed-form answer.
// ---------------------------------------------------------------------------
__global__ __launch_bounds__(256)
void sparsemax_rows_1024(float* __restrict__ z) {
    __shared__ float sm[32];
    const int D = 1024;
    float* zr = z + (size_t)blockIdx.x * D;

    float v[4];
#pragma unroll
    for (int i = 0; i < 4; ++i) v[i] = zr[threadIdx.x + i * 256];

    float mx = fmaxf(fmaxf(v[0], v[1]), fmaxf(v[2], v[3]));
    mx = block_max(mx, sm);

    float lo = mx - 1.0f, hi = mx;
    for (int it = 0; it < 32; ++it) {
        const float mid = 0.5f * (lo + hi);
        float s = 0.0f;
#pragma unroll
        for (int i = 0; i < 4; ++i) s += fmaxf(v[i] - mid, 0.0f);
        s = block_sum(s, sm);
        if (s > 1.0f) lo = mid; else hi = mid;   // uniform branch
    }
    float tau = 0.5f * (lo + hi);

    float cnt = 0.0f, sum = 0.0f;
#pragma unroll
    for (int i = 0; i < 4; ++i) {
        if (v[i] > tau) { cnt += 1.0f; sum += v[i]; }
    }
    cnt = block_sum(cnt, sm);
    sum = block_sum(sum, sm);
    tau = (sum - 1.0f) / fmaxf(cnt, 1.0f);

#pragma unroll
    for (int i = 0; i < 4; ++i)
        zr[threadIdx.x + i * 256] = fmaxf(v[i] - tau, 0.0f);
}

// ---------------------------------------------------------------------------
// LayerNorm over rows of length 512 (blockDim 256, 2 elems/thread).
// ---------------------------------------------------------------------------
__global__ __launch_bounds__(256)
void layernorm_rows_512(const float* __restrict__ x, const float* __restrict__ w,
                        const float* __restrict__ b, float* __restrict__ out) {
    __shared__ float sm[32];
    const int D = 512;
    const float* xr = x + (size_t)blockIdx.x * D;
    const float v0 = xr[threadIdx.x];
    const float v1 = xr[threadIdx.x + 256];

    const float mu  = block_sum(v0 + v1, sm) * (1.0f / D);
    const float d0  = v0 - mu, d1 = v1 - mu;
    const float var = block_sum(d0 * d0 + d1 * d1, sm) * (1.0f / D);
    const float rs  = rsqrtf(var + 1e-5f);

    float* orow = out + (size_t)blockIdx.x * D;
    orow[threadIdx.x]       = d0 * rs * w[threadIdx.x]       + b[threadIdx.x];
    orow[threadIdx.x + 256] = d1 * rs * w[threadIdx.x + 256] + b[threadIdx.x + 256];
}

// ---------------------------------------------------------------------------
// Host orchestration.
// ---------------------------------------------------------------------------
extern "C" void kernel_launch(void* const* d_in, const int* in_sizes, int n_in,
                              void* d_out, int out_size, void* d_ws, size_t ws_size,
                              hipStream_t stream) {
    (void)in_sizes; (void)n_in; (void)out_size; (void)ws_size;

    const int S = 1024, Bt = 4, E = 512, H = 8, F = 2048;
    const int M = S * Bt;             // 4096 token rows
    const int E3 = 3 * E;             // 1536

    const float* src        = (const float*)d_in[0];
    const float* in_proj_w  = (const float*)d_in[1];
    const float* in_proj_b  = (const float*)d_in[2];
    const float* out_proj_w = (const float*)d_in[3];
    const float* out_proj_b = (const float*)d_in[4];
    const float* lin1_w     = (const float*)d_in[5];
    const float* lin1_b     = (const float*)d_in[6];
    const float* lin2_w     = (const float*)d_in[7];
    const float* lin2_b     = (const float*)d_in[8];
    const float* norm1_w    = (const float*)d_in[9];
    const float* norm1_b    = (const float*)d_in[10];
    const float* norm2_w    = (const float*)d_in[11];
    const float* norm2_b    = (const float*)d_in[12];

    // Workspace layout (floats); regions reused once dead:
    float* ws     = (float*)d_ws;
    float* qkv    = ws;                                  // M*3E
    float* scores = qkv + (size_t)M * E3;                // B*H*S*S
    float* ctx    = scores + (size_t)Bt * H * S * S;     // M*E
    float* x      = ctx + (size_t)M * E;                 // M*E
    float* y1     = qkv;     // qkv dead after ctx
    float* ff     = scores;  // scores dead after ctx
    float* y2     = ctx;     // ctx dead after y1

    const dim3 blk(256);

    // 1) qkv = src @ in_proj_w.T + in_proj_b          [M, 3E]
    gemm_bf16_wmma<<<dim3(M / 128, E3 / 64, 1), blk, 0, stream>>>(
        src, 0, 0, E,
        in_proj_w, 0, 0, E,
        qkv, 0, 0, E3,
        in_proj_b, nullptr, E, 1.0f, 1, 0, 0);

    // 2) scores[b,h] = (Q * hd^-0.5 / 1.3) @ K^T     [S, S] per (b,h)
    const float alphaS = 0.125f / 1.3f;   // hd^-0.5 / SPARSEMAX_TEMP
    gemm_bf16_wmma<<<dim3(S / 128, S / 64, Bt * H), blk, 0, stream>>>(
        qkv,      E3, 64, Bt * E3,
        qkv + E,  E3, 64, Bt * E3,
        scores, (long long)H * S * S, (long long)S * S, S,
        nullptr, nullptr, 64, alphaS, H, 0, 0);

    // 3) sparsemax rows (in place on scores)
    sparsemax_rows_1024<<<dim3(Bt * H * S), dim3(256), 0, stream>>>(scores);

    // 4) ctx[s,b,h*64+d] = attn[b,h] @ V[b,h]         (B is k-major: V[t][d])
    gemm_bf16_wmma<<<dim3(S / 128, 64 / 64, Bt * H), blk, 0, stream>>>(
        scores, (long long)H * S * S, (long long)S * S, S,
        qkv + 2 * E, E3, 64, Bt * E3,
        ctx, E, 64, Bt * E,
        nullptr, nullptr, S, 1.0f, H, 1, 0);

    // 5) y1 = ctx @ out_proj_w.T + out_proj_b + src
    gemm_bf16_wmma<<<dim3(M / 128, E / 64, 1), blk, 0, stream>>>(
        ctx, 0, 0, E,
        out_proj_w, 0, 0, E,
        y1, 0, 0, E,
        out_proj_b, src, E, 1.0f, 1, 0, 0);

    // 6) x = LayerNorm(y1; norm1)
    layernorm_rows_512<<<dim3(M), dim3(256), 0, stream>>>(y1, norm1_w, norm1_b, x);

    // 7) ff = relu(x @ lin1_w.T + lin1_b)             [M, F]
    gemm_bf16_wmma<<<dim3(M / 128, F / 64, 1), blk, 0, stream>>>(
        x, 0, 0, E,
        lin1_w, 0, 0, E,
        ff, 0, 0, F,
        lin1_b, nullptr, E, 1.0f, 1, 0, 1);

    // 8) y2 = ff @ lin2_w.T + lin2_b + x
    gemm_bf16_wmma<<<dim3(M / 128, E / 64, 1), blk, 0, stream>>>(
        ff, 0, 0, F,
        lin2_w, 0, 0, F,
        y2, 0, 0, E,
        lin2_b, x, F, 1.0f, 1, 0, 0);

    // 9) out = LayerNorm(y2; norm2)
    layernorm_rows_512<<<dim3(M), dim3(256), 0, stream>>>(y2, norm2_w, norm2_b,
                                                          (float*)d_out);
}
